// WordLoss_20340965113904
// MI455X (gfx1250) — matrix operand order repaired and checked by reading, not compiled, not run on hardware
//
#include <hip/hip_runtime.h>
#include <hip/hip_bf16.h>
#include <math.h>

// ---------------- problem constants ----------------
#define BB  64          // batch
#define DD  256         // feature dim
#define TT  24          // words
#define RR  17
#define SS  (RR*RR)     // 289 spatial
#define SP  304         // S padded to 19*16 (M padding, GEMM1)
#define SK  320         // S padded to 10*32 (K padding, GEMM2)
#define TN  32          // T padded to 2*16 / 32 (N padding)
#define KB1 (DD/32)     // 8  k-blocks in GEMM1
#define KB2 (SK/32)     // 10 k-blocks in GEMM2
#define G1  4.0f
#define G2  5.0f
#define G3  10.0f
#define EPSI 1e-8f
#define NEGV (-1e30f)

typedef __attribute__((ext_vector_type(16))) _Float16 v16h;
typedef __attribute__((ext_vector_type(8)))  _Float16 v8h;
typedef __attribute__((ext_vector_type(8)))  float    v8f;

// ---------------- WMMA fragment helpers (contiguous k order for A/B, wave32) -------

// A fragment: 16x32 f16, row-major source with stride lda.
// lanes 0-15:  M=lane,    K = k0+{0..15}
// lanes 16-31: M=lane-16, K = k0+{16..31}
__device__ __forceinline__ v16h load_a_frag(const _Float16* A, int lda,
                                            int m0, int k0, int lane) {
    int row = m0 + (lane & 15);
    int kb  = k0 + ((lane >> 4) << 4);
    const _Float16* p = A + (size_t)row * lda + kb;
    v8h lo = *(const v8h*)p;
    v8h hi = *(const v8h*)(p + 8);
    v16h a;
#pragma unroll
    for (int j = 0; j < 8; ++j) { a[j] = lo[j]; a[8 + j] = hi[j]; }
    return a;
}

// Packed B fragment: operand pre-swizzled into per-lane register order.
// Fragment (kb, nt): offset = ((kb*2 + nt)*32 + lane)*16, 16 contiguous halves.
__device__ __forceinline__ v16h load_b_packed(const _Float16* Bp,
                                              int kb, int nt, int lane) {
    return *(const v16h*)(Bp + (((size_t)kb * 2 + nt) * 32 + lane) * 16);
}

// B-fragment scatter index for element (K=s, N=t):
// lane = (s%32 >= 16 ? 16 : 0) + (t&15), elem = s&15.
__device__ __forceinline__ int b_frag_index(int s, int t) {
    int kb   = s >> 5;
    int sr   = s & 31;
    int nt   = t >> 4;
    int lane = (sr & 16) + (t & 15);
    return (((kb * 2) + nt) * 32 + lane) * 16 + (sr & 15);
}

// C store: 16x16 f32. VGPR j: lanes 0-15 -> M=m0+j, N=n0+lane;
//                     lanes 16-31 -> M=m0+8+j, N=n0+lane-16.
__device__ __forceinline__ void store_c(float* Cm, int ldc,
                                        int m0, int n0, int lane, v8f c) {
    int col   = n0 + (lane & 15);
    int rbase = m0 + ((lane >> 4) << 3);
#pragma unroll
    for (int j = 0; j < 8; ++j) Cm[(size_t)(rbase + j) * ldc + col] = c[j];
}

__device__ __forceinline__ float wave_max(float v) {
#pragma unroll
    for (int o = 16; o > 0; o >>= 1) v = fmaxf(v, __shfl_xor(v, o, 32));
    return v;
}
__device__ __forceinline__ float wave_sum(float v) {
#pragma unroll
    for (int o = 16; o > 0; o >>= 1) v += __shfl_xor(v, o, 32);
    return v;
}

// ---------------- conversion kernels (f32 -> padded f16) -------------------

// ctx1[b][sp][d] = img[b][d][sp]  (sp < S else 0), transpose for GEMM1-A
__global__ void conv_ctx1(const float* __restrict__ img, _Float16* __restrict__ out) {
    size_t idx = (size_t)blockIdx.x * 256 + threadIdx.x;
    if (idx >= (size_t)BB * SP * DD) return;
    int d  = (int)(idx % DD);
    size_t r = idx / DD;
    int sp = (int)(r % SP);
    int b  = (int)(r / SP);
    float v = (sp < SS) ? img[((size_t)b * DD + d) * SS + sp] : 0.f;
    out[idx] = (_Float16)v;
}

// ctx2[b][d][sk] = img[b][d][sk]  (sk < S else 0), K-padded for GEMM2-A
__global__ void conv_ctx2(const float* __restrict__ img, _Float16* __restrict__ out) {
    size_t idx = (size_t)blockIdx.x * 256 + threadIdx.x;
    if (idx >= (size_t)BB * DD * SK) return;
    int sk = (int)(idx % SK);
    size_t r = idx / SK;
    float v = (sk < SS) ? img[r * SS + sk] : 0.f;
    out[idx] = (_Float16)v;
}

// words pre-swizzled into WMMA B-fragment layout:
// out[i][kb][nt][lane][e] = words[i][d = kb*32 + (lane>>4)*16 + e]
//                                 [t = nt*16 + (lane&15)]  (t < T else 0)
__global__ void conv_words_frag(const float* __restrict__ w, _Float16* __restrict__ out) {
    size_t idx = (size_t)blockIdx.x * 256 + threadIdx.x;     // B*KB1*2*32*16
    if (idx >= (size_t)BB * KB1 * 2 * 32 * 16) return;
    int e    = (int)(idx & 15);
    int lane = (int)((idx >> 4) & 31);
    int nt   = (int)((idx >> 9) & 1);
    int kb   = (int)((idx >> 10) & 7);
    int i    = (int)(idx >> 13);
    int d = kb * 32 + ((lane >> 4) << 4) + e;
    int t = nt * 16 + (lane & 15);
    float v = (t < TT) ? w[((size_t)i * DD + d) * TT + t] : 0.f;
    out[idx] = (_Float16)v;
}

// ---------------- fused per-(i,b) attention kernel -------------------------

__global__ __launch_bounds__(256) void pair_kernel(
    const _Float16* __restrict__ ctx_h1,   // [B][SP][D]
    const _Float16* __restrict__ ctx_h2,   // [B][D][SK]
    const _Float16* __restrict__ words_h,  // [B][KB1*2*512] fragment layout
    const float*    __restrict__ words_f,  // [B][D][T] original f32
    const int*      __restrict__ cap_lens,
    float* __restrict__ sim_ws,            // [B][B] = similarities[b][i]
    float* __restrict__ att_out)           // [B][T][S] diagonal maps
{
    const int bb = blockIdx.x;   // ctx index
    const int ii = blockIdx.y;   // words index
    const int tid  = threadIdx.x;
    const int lane = tid & 31;
    const int wv   = tid >> 5;   // 8 waves

    __shared__ float    sA[SP * TN];          // attn f32; later aliased as wei f32
    __shared__ _Float16 sB2[KB2 * 2 * 512];   // attn2 f16, B-fragment layout
    __shared__ float    sE[TT];

    // zero attn2 fragments (covers K padding s>=289 and N padding t>=24)
    for (int idx = tid; idx < KB2 * 2 * 512; idx += 256) sB2[idx] = (_Float16)0.f;

    // ---- GEMM1: attn[s,t] = sum_d ctx[d,s] * w[d,t] ----
    const _Float16* A1 = ctx_h1 + (size_t)bb * SP * DD;
    const _Float16* B1 = words_h + (size_t)ii * (KB1 * 2 * 512);
    for (int job = wv; job < (SP / 16) * (TN / 16); job += 8) {
        int m0 = (job >> 1) * 16;
        int nt = (job & 1);
        v8f c = {};
#pragma unroll 2
        for (int kb = 0; kb < KB1; ++kb) {
            v16h a = load_a_frag(A1, DD, m0, kb * 32, lane);
            v16h b = load_b_packed(B1, kb, nt, lane);
            c = __builtin_amdgcn_wmma_f32_16x16x32_f16(false, a, false, b,
                                                       (short)0, c, false, false);
        }
        store_c(sA, TN, m0, nt * 16, lane, c);
    }
    __syncthreads();

    const int len = cap_lens[ii];

    // ---- softmax over t (masked), per spatial row s ----
    for (int s = tid; s < SS; s += 256) {
        float m = NEGV;
        for (int t = 0; t < TT; ++t) {
            float v = (t < len) ? sA[s * TN + t] : NEGV;
            m = fmaxf(m, v);
        }
        float sum = 0.f;
        for (int t = 0; t < TT; ++t) {
            float v = (t < len) ? sA[s * TN + t] : NEGV;
            sum += __expf(v - m);
        }
        float inv = 1.f / sum;
        for (int t = 0; t < TT; ++t) {
            float v = (t < len) ? sA[s * TN + t] : NEGV;
            sA[s * TN + t] = __expf(v - m) * inv;   // masked -> 0
        }
    }
    __syncthreads();

    // ---- transpose * G1, softmax over s; scatter attn2 into B-fragment
    //      layout + emit diagonal att_maps ----
    for (int t = wv; t < TT; t += 8) {
        float m = NEGV;
        for (int s = lane; s < SS; s += 32) m = fmaxf(m, sA[s * TN + t]);
        m = wave_max(m) * G1;
        float sum = 0.f;
        for (int s = lane; s < SS; s += 32) sum += __expf(G1 * sA[s * TN + t] - m);
        sum = wave_sum(sum);
        float inv = 1.f / sum;
        for (int s = lane; s < SS; s += 32) {
            float p = __expf(G1 * sA[s * TN + t] - m) * inv;
            sB2[b_frag_index(s, t)] = (_Float16)p;
            if (ii == bb) att_out[((size_t)bb * TT + t) * SS + s] = p;
        }
    }
    __syncthreads();

    // ---- GEMM2: wei[d,t] = sum_s ctx[d,s] * attn2[t,s] (sA reused as wei)
    const _Float16* A2 = ctx_h2 + (size_t)bb * DD * SK;
    for (int job = wv; job < (DD / 16) * (TN / 16); job += 8) {
        int m0 = (job >> 1) * 16;
        int nt = (job & 1);
        v8f c = {};
#pragma unroll 2
        for (int kb = 0; kb < KB2; ++kb) {
            v16h a = load_a_frag(A2, SK, m0, kb * 32, lane);
            v16h b = load_b_packed(sB2, kb, nt, lane);
            c = __builtin_amdgcn_wmma_f32_16x16x32_f16(false, a, false, b,
                                                       (short)0, c, false, false);
        }
        store_c(sA, TN, m0, nt * 16, lane, c);   // wei[d][t]
    }
    __syncthreads();

    // ---- cosine similarity per t, masked exp ----
    const float* wf = words_f + (size_t)ii * DD * TT;
    for (int t = wv; t < TT; t += 8) {
        float num = 0.f, nw = 0.f, nv = 0.f;
        for (int d = lane; d < DD; d += 32) {
            float w = wf[(size_t)d * TT + t];
            float v = sA[d * TN + t];
            num += w * v; nw += w * w; nv += v * v;
        }
        num = wave_sum(num); nw = wave_sum(nw); nv = wave_sum(nv);
        if (lane == 0) {
            float den = fmaxf(sqrtf(nw) * sqrtf(nv), EPSI);
            float rs  = num / den;
            sE[t] = (t < len) ? __expf(G2 * rs) : 0.f;
        }
    }
    __syncthreads();

    if (tid == 0) {
        float s = 0.f;
        for (int t = 0; t < TT; ++t) s += sE[t];
        // similarities = sims.T * G3  -> row = bb, col = ii
        sim_ws[(size_t)bb * BB + ii] = G3 * logf(s);
    }
}

// ---------------- loss finalization (1 block, 64 threads) ------------------

__global__ void finalize_kernel(const float* __restrict__ sim,
                                const int* __restrict__ labels,
                                const int* __restrict__ class_ids,
                                float* __restrict__ out)
{
    __shared__ float r0[BB], r1[BB];
    const int r  = threadIdx.x;
    const int cr = class_ids[r];
    const int l  = labels[r];

    // loss0: log_softmax over row r of masked similarities
    float m = NEGV;
    for (int c = 0; c < BB; ++c) {
        float v = (class_ids[c] == cr && c != r) ? NEGV : sim[(size_t)r * BB + c];
        m = fmaxf(m, v);
    }
    float sum = 0.f;
    for (int c = 0; c < BB; ++c) {
        float v = (class_ids[c] == cr && c != r) ? NEGV : sim[(size_t)r * BB + c];
        sum += __expf(v - m);
    }
    float vl = (class_ids[l] == cr && l != r) ? NEGV : sim[(size_t)r * BB + l];
    r0[r] = vl - m - logf(sum);

    // loss1: log_softmax over column r (row r of transpose)
    m = NEGV;
    for (int c = 0; c < BB; ++c) {
        float v = (class_ids[r] == class_ids[c] && c != r) ? NEGV : sim[(size_t)c * BB + r];
        m = fmaxf(m, v);
    }
    sum = 0.f;
    for (int c = 0; c < BB; ++c) {
        float v = (class_ids[r] == class_ids[c] && c != r) ? NEGV : sim[(size_t)c * BB + r];
        sum += __expf(v - m);
    }
    float vl1 = (class_ids[l] == cr && l != r) ? NEGV : sim[(size_t)l * BB + r];
    r1[r] = vl1 - m - logf(sum);

    __syncthreads();
    if (r == 0) {
        float s0 = 0.f, s1 = 0.f;
        for (int i = 0; i < BB; ++i) { s0 += r0[i]; s1 += r1[i]; }
        out[0] = -s0 / BB;
        out[1] = -s1 / BB;
    }
}

// ---------------- host launch ----------------------------------------------

extern "C" void kernel_launch(void* const* d_in, const int* in_sizes, int n_in,
                              void* d_out, int out_size, void* d_ws, size_t ws_size,
                              hipStream_t stream) {
    const float* img   = (const float*)d_in[0];   // (B,D,R,R)
    const float* words = (const float*)d_in[1];   // (B,D,T)
    const int* labels    = (const int*)d_in[2];
    const int* cap_lens  = (const int*)d_in[3];
    const int* class_ids = (const int*)d_in[4];
    float* out = (float*)d_out;                   // [loss0, loss1, att_maps...]

    // workspace carve-up (~21.5 MB)
    _Float16* ctx_h1  = (_Float16*)d_ws;                  // B*SP*D
    _Float16* ctx_h2  = ctx_h1 + (size_t)BB * SP * DD;    // B*D*SK
    _Float16* words_h = ctx_h2 + (size_t)BB * DD * SK;    // B*KB1*2*512
    float*    sim_ws  = (float*)(words_h + (size_t)BB * KB1 * 2 * 512);

    {
        size_t n = (size_t)BB * SP * DD;
        conv_ctx1<<<dim3((unsigned)((n + 255) / 256)), 256, 0, stream>>>(img, ctx_h1);
    }
    {
        size_t n = (size_t)BB * DD * SK;
        conv_ctx2<<<dim3((unsigned)((n + 255) / 256)), 256, 0, stream>>>(img, ctx_h2);
    }
    {
        size_t n = (size_t)BB * KB1 * 2 * 512;
        conv_words_frag<<<dim3((unsigned)((n + 255) / 256)), 256, 0, stream>>>(words, words_h);
    }

    pair_kernel<<<dim3(BB, BB), 256, 0, stream>>>(ctx_h1, ctx_h2, words_h,
                                                  words, cap_lens, sim_ws, out + 2);

    finalize_kernel<<<1, BB, 0, stream>>>(sim_ws, labels, class_ids, out);
}
